// PatchMLP_30030411333659
// MI455X (gfx1250) — matrix operand order repaired and checked by reading, not compiled, not run on hardware
//
#include <hip/hip_runtime.h>

// PatchMLP fused implicit-GEMM for MI455X (gfx1250, wave32).
// conv(50x50,128) as GEMM [pixels x 3200] x [3200 x 128] with
// v_wmma_f32_16x16x32_f16. W1 K-tiles are triple-buffered into LDS via
// global_load_async_to_lds_b128 (distance-2 prefetch, s_wait_asynccnt),
// rows padded to 80B for bank-conflict-free ds_load_b128. Fused layer2
// WMMA through an LDS transpose and layer3 GEMV with shfl_xor reduction.

typedef _Float16 v16h __attribute__((ext_vector_type(16)));
typedef _Float16 v8h  __attribute__((ext_vector_type(8)));
typedef float    v8f  __attribute__((ext_vector_type(8)));

#define XH 305        // padded rows: 256 + 49
#define XW 320        // padded row stride
#define KROW 64       // patch row padded 50 -> 64 (two K=32 chunks)
#define KTOT 3200     // 50 * 64
#define HID 128
#define NSTEP 100     // 50 py * 2 kc
#define TROWB 80      // staged tile row stride in bytes (64B data + 16B pad)
#define TILEB (HID * TROWB)  // 10240 B per tile buffer

// ---- prep: zero-padded f16 halo image -------------------------------------
__global__ void pad_x_kernel(const float* __restrict__ x,
                             _Float16* __restrict__ xpad) {
  int idx = blockIdx.x * blockDim.x + threadIdx.x;
  const int n = 2 * XH * XW;
  if (idx >= n) return;
  int b   = idx / (XH * XW);
  int rem = idx - b * (XH * XW);
  int r   = rem / XW;
  int c   = rem - r * XW;
  int rr = r - 25, cc = c - 25;
  float v = 0.0f;
  if (rr >= 0 && rr < 256 && cc >= 0 && cc < 256)
    v = x[(b * 256 + rr) * 256 + cc];
  xpad[idx] = (_Float16)v;
}

// ---- prep: W1 [128,2500] f32 -> [128][3200] f16, patch rows padded 50->64 --
__global__ void pack_w1_kernel(const float* __restrict__ W1,
                               _Float16* __restrict__ w1p) {
  int idx = blockIdx.x * blockDim.x + threadIdx.x;
  const int n = HID * KTOT;
  if (idx >= n) return;
  int ch = idx / KTOT;
  int k  = idx - ch * KTOT;
  int py = k >> 6;
  int px = k & 63;
  float v = (px < 50) ? W1[ch * 2500 + py * 50 + px] : 0.0f;
  w1p[idx] = (_Float16)v;
}

// ---- prep: W2 [128,128] f32 -> f16 ----------------------------------------
__global__ void pack_w2_kernel(const float* __restrict__ W2,
                               _Float16* __restrict__ w2p) {
  int idx = blockIdx.x * blockDim.x + threadIdx.x;
  if (idx >= HID * HID) return;
  w2p[idx] = (_Float16)W2[idx];
}

// ---- fused main ------------------------------------------------------------
__global__ __launch_bounds__(256) void patchmlp_fused(
    const _Float16* __restrict__ xpad, const _Float16* __restrict__ w1p,
    const _Float16* __restrict__ w2p,
    const float* __restrict__ b1, const float* __restrict__ b2,
    const float* __restrict__ w3, const float* __restrict__ b3,
    float* __restrict__ out) {
  // 64 KB LDS. During the conv K-loop the first 30 KB triple-buffers W1
  // K-tiles (3 x 128 rows x 80B); afterwards (with a barrier) the whole
  // region is reused as h1 [pixel][channel] f16.
  __shared__ _Float16 smem[32768];

  const int bi = blockIdx.x >> 8;   // batch
  const int y  = blockIdx.x & 255;  // image row
  const int w  = threadIdx.x >> 5;  // wave 0..7
  const int l  = threadIdx.x & 31;
  const int m  = l & 15;            // A: M-row / B,C: N-col
  const int hi = l >> 4;            // lane-half K/M split
  const int wm = w & 3;             // pixel group: wm*64 .. +63
  const int wn = w >> 2;            // channel group: wn*64 .. +63

  // ---- async staging setup: thread tt copies 32B of each W1 K-tile ----
  const int tt = threadIdx.x;
  const char* srow =
      (const char*)w1p + (size_t)(tt >> 1) * (KTOT * 2) + (size_t)((tt & 1) * 32);
  const unsigned dbase = (unsigned)(uintptr_t)(&smem[0]) +
                         (unsigned)((tt >> 1) * TROWB + (tt & 1) * 32);

#define STAGE_TILE(s, bufidx)                                                  \
  {                                                                            \
    unsigned dst_ = dbase + (unsigned)(bufidx) * (unsigned)TILEB;              \
    const char* src_ = srow + (size_t)(s) * 64;                                \
    asm volatile(                                                              \
        "global_load_async_to_lds_b128 %0, %1, off\n\t"                        \
        "global_load_async_to_lds_b128 %0, %1, off offset:16" ::"v"(dst_),     \
        "v"(src_)                                                              \
        : "memory");                                                           \
  }

  // ---- layer 1: implicit GEMM conv, wave tile = 64 pixels x 64 channels ----
  v8f acc[4][4];
#pragma unroll
  for (int i = 0; i < 4; ++i)
#pragma unroll
    for (int j = 0; j < 4; ++j)
      acc[i][j] = (v8f){0.f, 0.f, 0.f, 0.f, 0.f, 0.f, 0.f, 0.f};

  STAGE_TILE(0, 0);
  STAGE_TILE(1, 1);
  for (int s = 0; s < NSTEP; ++s) {
    // drain tile s (2 async ops may remain in flight for tile s+1)
    if (s < NSTEP - 1) {
      asm volatile("s_wait_asynccnt 0x2" ::: "memory");
    } else {
      asm volatile("s_wait_asynccnt 0x0" ::: "memory");
    }
    __syncthreads();  // tile s resident everywhere; buffer (s+2)%3 free
    if (s + 2 < NSTEP) STAGE_TILE(s + 2, (s + 2) % 3);

    const int py = s >> 1;
    const int kc = s & 1;
    const _Float16* xrow = xpad + ((size_t)bi * XH + (y + py)) * XW;

    // A groups: 5 runs of 8 consecutive halves feed 4 overlapping fragments.
    // layout (16-bit A 16x32): e<8 -> K=hi*8+e ; e>=8 -> K=16+hi*8+(e-8)
    const int c0 = wm * 64 + m + kc * 32 + hi * 8;
    _Float16 ag[5][8];
#pragma unroll
    for (int t = 0; t < 5; ++t)
#pragma unroll
      for (int e = 0; e < 8; ++e) ag[t][e] = xrow[c0 + 16 * t + e];

    v16h afr[4];
#pragma unroll
    for (int ms = 0; ms < 4; ++ms)
#pragma unroll
      for (int e = 0; e < 8; ++e) {
        afr[ms][e]     = ag[ms][e];
        afr[ms][8 + e] = ag[ms + 1][e];
      }

    // B from LDS tile (row stride 80B = 20 DW -> conflict-free across lanes):
    // lane n=m, element e<8 -> K = hi*8 + e ; e>=8 -> K = 16 + hi*8 + (e-8)
    const _Float16* bufp = smem + (s % 3) * (TILEB / 2);
    v16h bfr[4];
#pragma unroll
    for (int ns = 0; ns < 4; ++ns)
    {
      const _Float16* bp_ = bufp + (wn * 64 + ns * 16 + m) * (TROWB / 2) + hi * 8;
      v8h blo_ = *(const v8h*)bp_;
      v8h bhi_ = *(const v8h*)(bp_ + 16);
      bfr[ns] = __builtin_shufflevector(blo_, bhi_, 0, 1, 2, 3, 4, 5, 6, 7, 8, 9,
                                        10, 11, 12, 13, 14, 15);
    }
#pragma unroll
    for (int ns = 0; ns < 4; ++ns)
#pragma unroll
      for (int ms = 0; ms < 4; ++ms)
        acc[ms][ns] = __builtin_amdgcn_wmma_f32_16x16x32_f16(
            false, afr[ms], false, bfr[ns], (short)0, acc[ms][ns], false, false);
  }
  __syncthreads();  // all tile consumption done -> safe to reuse LDS as h1

  // ---- +b1, f32->f16, transpose into LDS as [pixel][channel] ----
  // C layout: VGPR r, lane -> pixel = base + r + 8*hi, channel = col + m
  _Float16* h1s = smem;
#pragma unroll
  for (int ms = 0; ms < 4; ++ms) {
#pragma unroll
    for (int ns = 0; ns < 4; ++ns) {
      float bb = b1[wn * 64 + ns * 16 + m];
      int prow = wm * 64 + ms * 16 + hi * 8;
#pragma unroll
      for (int r = 0; r < 8; ++r)
        h1s[(prow + r) * HID + wn * 64 + ns * 16 + m] =
            (_Float16)(acc[ms][ns][r] + bb);
    }
  }
  __syncthreads();

  // ---- layer 2: M=out-channels (W2 as A), N=pixels (h1 from LDS as B) ----
  // wave w handles pixels w*32 .. w*32+31 (2 N-subtiles), all 128 out-ch.
  v8f acc2[2][8];
#pragma unroll
  for (int i = 0; i < 2; ++i)
#pragma unroll
    for (int j = 0; j < 8; ++j)
      acc2[i][j] = (v8f){0.f, 0.f, 0.f, 0.f, 0.f, 0.f, 0.f, 0.f};

#pragma unroll
  for (int kc2 = 0; kc2 < 4; ++kc2) {
    v16h bf0 = *(const v16h*)&h1s[(w * 32 + m) * HID + kc2 * 32 + hi * 16];
    v16h bf1 = *(const v16h*)&h1s[(w * 32 + 16 + m) * HID + kc2 * 32 + hi * 16];
#pragma unroll
    for (int ms2 = 0; ms2 < 8; ++ms2) {
      const _Float16* ap =
          w2p + (size_t)(ms2 * 16 + m) * HID + kc2 * 32 + hi * 16;
      v8h alo = *(const v8h*)ap;         // K = kc2*32 + hi*16 + 0..7
      v8h ahi = *(const v8h*)(ap + 8);   // K = kc2*32 + hi*16 + 8..15
      v16h af = __builtin_shufflevector(alo, ahi, 0, 1, 2, 3, 4, 5, 6, 7, 8, 9,
                                        10, 11, 12, 13, 14, 15);
      acc2[0][ms2] = __builtin_amdgcn_wmma_f32_16x16x32_f16(
          false, af, false, bf0, (short)0, acc2[0][ms2], false, false);
      acc2[1][ms2] = __builtin_amdgcn_wmma_f32_16x16x32_f16(
          false, af, false, bf1, (short)0, acc2[1][ms2], false, false);
    }
  }

  // ---- +b2, relu, layer3 (2x128 GEMV) in registers ----
  // acc2: VGPR r, lane -> out-ch = ms2*16 + r + 8*hi, pixel = base + m
  float o[2][2] = {{0.f, 0.f}, {0.f, 0.f}};  // [ps][class] lane-half partials
#pragma unroll
  for (int ms2 = 0; ms2 < 8; ++ms2) {
    const float* b2p  = b2 + ms2 * 16 + hi * 8;
    const float* w3p0 = w3 + ms2 * 16 + hi * 8;
    const float* w3p1 = w3 + HID + ms2 * 16 + hi * 8;
#pragma unroll
    for (int r = 0; r < 8; ++r) {
      float bb  = b2p[r];
      float w0  = w3p0[r];
      float w1v = w3p1[r];
#pragma unroll
      for (int ps = 0; ps < 2; ++ps) {
        float h = acc2[ps][ms2][r] + bb;
        h = h > 0.f ? h : 0.f;
        o[ps][0] += w0 * h;
        o[ps][1] += w1v * h;
      }
    }
  }
#pragma unroll
  for (int ps = 0; ps < 2; ++ps) {
#pragma unroll
    for (int cls = 0; cls < 2; ++cls) {
      float t = o[ps][cls];
      t += __shfl_xor(t, 16, 32);   // combine lane-half channel partials
      t += b3[cls];
      if (hi == 0) {
        out[(((size_t)bi * 2 + cls) * 256 + y) * 256 + (w * 32 + ps * 16 + m)] =
            t;
      }
    }
  }
#undef STAGE_TILE
}

extern "C" void kernel_launch(void* const* d_in, const int* in_sizes, int n_in,
                              void* d_out, int out_size, void* d_ws,
                              size_t ws_size, hipStream_t stream) {
  (void)in_sizes; (void)n_in; (void)out_size; (void)ws_size;
  const float* x  = (const float*)d_in[0];
  const float* W1 = (const float*)d_in[1];
  const float* b1 = (const float*)d_in[2];
  const float* W2 = (const float*)d_in[3];
  const float* b2 = (const float*)d_in[4];
  const float* W3 = (const float*)d_in[5];
  const float* b3 = (const float*)d_in[6];
  float* out = (float*)d_out;

  // workspace layout (offsets 32B-aligned), total ~1.22 MB
  char* ws = (char*)d_ws;
  _Float16* xpad = (_Float16*)(ws);                      // 2*305*320*2 = 390400 B
  _Float16* w1p  = (_Float16*)(ws + 390400);             // 128*3200*2  = 819200 B
  _Float16* w2p  = (_Float16*)(ws + 390400 + 819200);    // 128*128*2   =  32768 B

  const int npadx = 2 * XH * XW;  // 195200
  const int nw1   = HID * KTOT;   // 409600
  const int nw2   = HID * HID;    // 16384
  pad_x_kernel<<<(npadx + 255) / 256, 256, 0, stream>>>(x, xpad);
  pack_w1_kernel<<<(nw1 + 255) / 256, 256, 0, stream>>>(W1, w1p);
  pack_w2_kernel<<<(nw2 + 255) / 256, 256, 0, stream>>>(W2, w2p);

  patchmlp_fused<<<512, 256, 0, stream>>>(xpad, w1p, w2p, b1, b2, W3, b3, out);
}
